// Decoder_28046136443315
// MI455X (gfx1250) — compile-verified, matrix operand order checked by reading, not run on hardware
//
#include <hip/hip_runtime.h>
#include <cmath>

// ---------------------------------------------------------------------------
// CDNA5 (gfx1250) transformer decoder. wave32, WMMA bf16, LDS fragment-layout.
// ---------------------------------------------------------------------------

typedef __attribute__((ext_vector_type(16))) __bf16 v16bf;
typedef __attribute__((ext_vector_type(8)))  __bf16 v8bf;
typedef __attribute__((ext_vector_type(8)))  float  v8f;

#define D_MODEL 1024
#define DFF     4096
#define SEQ     512
#define BATCH   4
#define HEADS   16
#define DEPTH   64
#define NTOK    (BATCH * SEQ)          // 2048 tokens
#define BH      (BATCH * HEADS)        // 64 head-batches

#define BM 128                          // block tile M
#define BN 64                           // block tile N
#define BK 64                           // staged K chunk (2 wmma K-steps)
#define LDA 72                          // bf16 row stride: 144B = 16B-aligned rows
#define LDB 72

__device__ __forceinline__ __bf16 tobf(float x) { return (__bf16)x; }

// Load a 16-element WMMA fragment as two aligned 16B runs -> 2x ds_load_b128.
// p must point at (row base + ks*32 + half*8); rows are 16B aligned.
__device__ __forceinline__ v16bf frag_ld(const __bf16* __restrict__ p)
{
    const v8bf lo = *reinterpret_cast<const v8bf*>(p);        // K = half*8 + 0..7
    const v8bf hi = *reinterpret_cast<const v8bf*>(p + 16);   // K = 16 + half*8 + 0..7
    return __builtin_shufflevector(lo, hi, 0, 1, 2, 3, 4, 5, 6, 7,
                                   8, 9, 10, 11, 12, 13, 14, 15);
}

__device__ __forceinline__ v8f wmma_bf16(v16bf a, v16bf b, v8f c)
{
    return __builtin_amdgcn_wmma_f32_16x16x32_bf16(false, a, false, b,
                                                   (short)0, c, false, false);
}

// float4 -> 4 contiguous bf16 in LDS (v_cvt_pk_bf16_f32 x2 + ds_store_b64).
__device__ __forceinline__ void st4(__bf16* __restrict__ d, float4 v)
{
    d[0] = tobf(v.x); d[1] = tobf(v.y); d[2] = tobf(v.z); d[3] = tobf(v.w);
}

// ---------------------------------------------------------------------------
// 1) Embedding gather + sqrt(D) scale + truncated sin/cos positional encoding
// ---------------------------------------------------------------------------
__global__ __launch_bounds__(256)
void embed_kernel(const int* __restrict__ tok,
                  const float* __restrict__ emb,
                  float* __restrict__ x)
{
    const int row = blockIdx.x;            // token index b*SEQ + s
    const int s   = row & (SEQ - 1);
    const int t   = threadIdx.x;
    const int id  = tok[row];
#pragma unroll
    for (int i = 0; i < 4; ++i) {
        const int c    = t + i * 256;
        const int even = c & ~1;
        const float ang = (float)s * powf(10000.0f, -(float)even / (float)D_MODEL);
        const float pe  = (c & 1) ? truncf(cosf(ang)) : truncf(sinf(ang));
        x[(size_t)row * D_MODEL + c] =
            emb[(size_t)id * D_MODEL + c] * 32.0f + pe;  // sqrt(1024) = 32
    }
}

// ---------------------------------------------------------------------------
// 2) GEMM: C[M,N] = A[M,K] * W[K,N] + bias[N]  (optional ReLU)
//    128x64 block tile, staged K=64, 8 waves (4x2), wave tile 32x32:
//    8 wmma per wave per stage off 2x(4+4) ds_load_b128.
// ---------------------------------------------------------------------------
template <int RELU>
__global__ __launch_bounds__(256)
void gemm_bias_kernel(const float* __restrict__ A,
                      const float* __restrict__ W,
                      const float* __restrict__ bias,
                      float* __restrict__ C,
                      int M, int N, int K)
{
    __shared__ __align__(16) __bf16 As[BM][LDA];   // A tile, row-major
    __shared__ __align__(16) __bf16 Bt[BN][LDB];   // B tile, TRANSPOSED: Bt[n][k]

    const int tid  = threadIdx.x;
    const int wave = tid >> 5;
    const int lane = tid & 31;
    const int wm   = wave & 3;          // 4 wave-rows of 32
    const int wn   = wave >> 2;         // 2 wave-cols of 32
    const int half = lane >> 4;
    const int r16  = lane & 15;
    const int row0 = blockIdx.y * BM;
    const int col0 = blockIdx.x * BN;

    v8f acc00 = {}, acc01 = {}, acc10 = {}, acc11 = {};

    for (int k0 = 0; k0 < K; k0 += BK) {
        // A tile: 128x64 f32 = 2048 float4, 8 per thread, fully coalesced.
#pragma unroll
        for (int i = 0; i < 8; ++i) {
            const int v = tid + i * 256;
            const int r = v >> 4, c4 = (v & 15) << 2;
            const float4 a4 = *reinterpret_cast<const float4*>(
                A + (size_t)(row0 + r) * K + k0 + c4);
            st4(&As[r][c4], a4);
        }
        // W tile: 64x64 f32 = 1024 float4, 4 per thread; transpose into Bt.
#pragma unroll
        for (int i = 0; i < 4; ++i) {
            const int v = tid + i * 256;
            const int r = v >> 4, c4 = (v & 15) << 2;   // r = k, c4 = n
            const float4 w4 = *reinterpret_cast<const float4*>(
                W + (size_t)(k0 + r) * N + col0 + c4);
            Bt[c4 + 0][r] = tobf(w4.x);
            Bt[c4 + 1][r] = tobf(w4.y);
            Bt[c4 + 2][r] = tobf(w4.z);
            Bt[c4 + 3][r] = tobf(w4.w);
        }
        if (k0 + BK < K) {   // prefetch next K-tile (global_prefetch_b8)
            __builtin_prefetch(A + (size_t)(row0 + (tid >> 4)) * K + k0 + BK + ((tid & 15) << 2), 0, 3);
            __builtin_prefetch(W + (size_t)(k0 + BK + (tid >> 4)) * N + col0 + ((tid & 15) << 2), 0, 3);
        }
        __syncthreads();

#pragma unroll
        for (int ks = 0; ks < 2; ++ks) {            // two 16x16x32 sub-steps
            const int kof = ks * 32 + half * 8;
            const v16bf fa0 = frag_ld(&As[wm * 32 + r16][kof]);
            const v16bf fa1 = frag_ld(&As[wm * 32 + 16 + r16][kof]);
            const v16bf fb0 = frag_ld(&Bt[wn * 32 + r16][kof]);
            const v16bf fb1 = frag_ld(&Bt[wn * 32 + 16 + r16][kof]);
            acc00 = wmma_bf16(fa0, fb0, acc00);
            acc01 = wmma_bf16(fa0, fb1, acc01);
            acc10 = wmma_bf16(fa1, fb0, acc10);
            acc11 = wmma_bf16(fa1, fb1, acc11);
        }
        __syncthreads();
    }

#pragma unroll
    for (int j = 0; j < 8; ++j) {      // C layout: VGPR j -> row j + 8*half
        const int m0 = row0 + wm * 32 + half * 8 + j;
        const int m1 = m0 + 16;
        const int n0 = col0 + wn * 32 + r16;
        float v00 = acc00[j] + bias[n0];
        float v01 = acc01[j] + bias[n0 + 16];
        float v10 = acc10[j] + bias[n0];
        float v11 = acc11[j] + bias[n0 + 16];
        if (RELU) {
            v00 = fmaxf(v00, 0.0f); v01 = fmaxf(v01, 0.0f);
            v10 = fmaxf(v10, 0.0f); v11 = fmaxf(v11, 0.0f);
        }
        C[(size_t)m0 * N + n0]      = v00;
        C[(size_t)m0 * N + n0 + 16] = v01;
        C[(size_t)m1 * N + n0]      = v10;
        C[(size_t)m1 * N + n0 + 16] = v11;
    }
}

// ---------------------------------------------------------------------------
// 3) Attention scores: S = Q.K^T / 8 + mask*(-1e9), per (b,h). Single stage
//    (depth=64). K^T operand is depth-contiguous -> contiguous LDS stores.
//    mask_mode 0: padding [B,1,1,S] (per b,k); 1: lookahead [S,S] (per q,k)
// ---------------------------------------------------------------------------
__global__ __launch_bounds__(256)
void attn_scores_kernel(const float* __restrict__ Q,
                        const float* __restrict__ Kact,
                        const float* __restrict__ mask,
                        int mask_mode,
                        float* __restrict__ scores)
{
    __shared__ __align__(16) __bf16 As[BM][LDA];
    __shared__ __align__(16) __bf16 Bt[BN][LDB];   // Bt[key][depth]

    const int tid  = threadIdx.x;
    const int wave = tid >> 5, lane = tid & 31;
    const int wm   = wave & 3, wn = wave >> 2;
    const int half = lane >> 4, r16 = lane & 15;
    const int bh = blockIdx.z;
    const int b  = bh / HEADS, h = bh % HEADS;
    const int q0 = blockIdx.y * BM;
    const int n0 = blockIdx.x * BN;

    v8f acc00 = {}, acc01 = {}, acc10 = {}, acc11 = {};

    // One stage: full depth 64.
#pragma unroll
    for (int i = 0; i < 8; ++i) {       // Q tile 128x64
        const int v = tid + i * 256;
        const int r = v >> 4, c4 = (v & 15) << 2;
        const float4 a4 = *reinterpret_cast<const float4*>(
            Q + (size_t)(b * SEQ + q0 + r) * D_MODEL + h * DEPTH + c4);
        st4(&As[r][c4], a4);
    }
#pragma unroll
    for (int i = 0; i < 4; ++i) {       // K tile 64 keys x 64 depth
        const int v = tid + i * 256;
        const int n = v >> 4, c4 = (v & 15) << 2;   // depth offset
        const float4 k4 = *reinterpret_cast<const float4*>(
            Kact + (size_t)(b * SEQ + n0 + n) * D_MODEL + h * DEPTH + c4);
        st4(&Bt[n][c4], k4);            // contiguous LDS store
    }
    __syncthreads();

#pragma unroll
    for (int ks = 0; ks < 2; ++ks) {
        const int kof = ks * 32 + half * 8;
        const v16bf fa0 = frag_ld(&As[wm * 32 + r16][kof]);
        const v16bf fa1 = frag_ld(&As[wm * 32 + 16 + r16][kof]);
        const v16bf fb0 = frag_ld(&Bt[wn * 32 + r16][kof]);
        const v16bf fb1 = frag_ld(&Bt[wn * 32 + 16 + r16][kof]);
        acc00 = wmma_bf16(fa0, fb0, acc00);
        acc01 = wmma_bf16(fa0, fb1, acc01);
        acc10 = wmma_bf16(fa1, fb0, acc10);
        acc11 = wmma_bf16(fa1, fb1, acc11);
    }

    const float scale = 0.125f;         // 1/sqrt(64)
#pragma unroll
    for (int mi = 0; mi < 2; ++mi) {
#pragma unroll
        for (int j = 0; j < 8; ++j) {
            const int m = q0 + wm * 32 + mi * 16 + half * 8 + j;
            const int n = n0 + wn * 32 + r16;
            const float a0 = (mi == 0) ? acc00[j] : acc10[j];
            const float a1 = (mi == 0) ? acc01[j] : acc11[j];
            float mk0, mk1;
            if (mask_mode == 0) { mk0 = mask[b * SEQ + n];          mk1 = mask[b * SEQ + n + 16]; }
            else                { mk0 = mask[(size_t)m * SEQ + n];  mk1 = mask[(size_t)m * SEQ + n + 16]; }
            scores[((size_t)bh * SEQ + m) * SEQ + n]      = a0 * scale + mk0 * (-1e9f);
            scores[((size_t)bh * SEQ + m) * SEQ + n + 16] = a1 * scale + mk1 * (-1e9f);
        }
    }
}

// ---------------------------------------------------------------------------
// 4) Row softmax over 512 elements, in place (one block per row).
// ---------------------------------------------------------------------------
__global__ __launch_bounds__(256)
void softmax_kernel(float* __restrict__ p)
{
    __shared__ float red[256];
    const int t = threadIdx.x;
    float* rp = p + (size_t)blockIdx.x * SEQ;

    const float a = rp[t];
    const float b = rp[t + 256];
    red[t] = fmaxf(a, b);
    __syncthreads();
#pragma unroll
    for (int s = 128; s > 0; s >>= 1) {
        if (t < s) red[t] = fmaxf(red[t], red[t + s]);
        __syncthreads();
    }
    const float mx = red[0];
    __syncthreads();

    const float e0 = expf(a - mx);
    const float e1 = expf(b - mx);
    red[t] = e0 + e1;
    __syncthreads();
#pragma unroll
    for (int s = 128; s > 0; s >>= 1) {
        if (t < s) red[t] += red[t + s];
        __syncthreads();
    }
    const float inv = 1.0f / red[0];
    rp[t]       = e0 * inv;
    rp[t + 256] = e1 * inv;
}

// ---------------------------------------------------------------------------
// 5) attn x V: O[b,q,h*64+d] = sum_k P[bh,q,k] * V[b,k,h*64+d]
//    128 q-rows x full depth-64 per block, staged K=64 over 512 keys.
// ---------------------------------------------------------------------------
__global__ __launch_bounds__(256)
void attn_v_kernel(const float* __restrict__ P,
                   const float* __restrict__ V,
                   float* __restrict__ O)
{
    __shared__ __align__(16) __bf16 As[BM][LDA];
    __shared__ __align__(16) __bf16 Bt[BN][LDB];   // Bt[depth][key]

    const int tid  = threadIdx.x;
    const int wave = tid >> 5, lane = tid & 31;
    const int wm   = wave & 3, wn = wave >> 2;
    const int half = lane >> 4, r16 = lane & 15;
    const int bh = blockIdx.z;
    const int b  = bh / HEADS, h = bh % HEADS;
    const int q0 = blockIdx.y * BM;

    v8f acc00 = {}, acc01 = {}, acc10 = {}, acc11 = {};

    for (int k0 = 0; k0 < SEQ; k0 += BK) {
#pragma unroll
        for (int i = 0; i < 8; ++i) {       // probs tile 128x64
            const int v = tid + i * 256;
            const int r = v >> 4, c4 = (v & 15) << 2;
            const float4 a4 = *reinterpret_cast<const float4*>(
                P + ((size_t)bh * SEQ + q0 + r) * SEQ + k0 + c4);
            st4(&As[r][c4], a4);
        }
#pragma unroll
        for (int i = 0; i < 4; ++i) {       // V tile 64 keys x 64 depth -> transpose
            const int v = tid + i * 256;
            const int r = v >> 4, c4 = (v & 15) << 2;   // r = key, c4 = depth
            const float4 v4 = *reinterpret_cast<const float4*>(
                V + (size_t)(b * SEQ + k0 + r) * D_MODEL + h * DEPTH + c4);
            Bt[c4 + 0][r] = tobf(v4.x);
            Bt[c4 + 1][r] = tobf(v4.y);
            Bt[c4 + 2][r] = tobf(v4.z);
            Bt[c4 + 3][r] = tobf(v4.w);
        }
        if (k0 + BK < SEQ) {
            __builtin_prefetch(P + ((size_t)bh * SEQ + q0 + (tid >> 4)) * SEQ + k0 + BK + ((tid & 15) << 2), 0, 3);
            __builtin_prefetch(V + (size_t)(b * SEQ + k0 + BK + (tid >> 4)) * D_MODEL + h * DEPTH + ((tid & 15) << 2), 0, 3);
        }
        __syncthreads();

#pragma unroll
        for (int ks = 0; ks < 2; ++ks) {
            const int kof = ks * 32 + half * 8;
            const v16bf fa0 = frag_ld(&As[wm * 32 + r16][kof]);
            const v16bf fa1 = frag_ld(&As[wm * 32 + 16 + r16][kof]);
            const v16bf fb0 = frag_ld(&Bt[wn * 32 + r16][kof]);
            const v16bf fb1 = frag_ld(&Bt[wn * 32 + 16 + r16][kof]);
            acc00 = wmma_bf16(fa0, fb0, acc00);
            acc01 = wmma_bf16(fa0, fb1, acc01);
            acc10 = wmma_bf16(fa1, fb0, acc10);
            acc11 = wmma_bf16(fa1, fb1, acc11);
        }
        __syncthreads();
    }

#pragma unroll
    for (int mi = 0; mi < 2; ++mi) {
#pragma unroll
        for (int j = 0; j < 8; ++j) {
            const int m = q0 + wm * 32 + mi * 16 + half * 8 + j;
            const int n = wn * 32 + r16;
            const float a0 = (mi == 0) ? acc00[j] : acc10[j];
            const float a1 = (mi == 0) ? acc01[j] : acc11[j];
            O[(size_t)(b * SEQ + m) * D_MODEL + h * DEPTH + n]      = a0;
            O[(size_t)(b * SEQ + m) * D_MODEL + h * DEPTH + n + 16] = a1;
        }
    }
}

// ---------------------------------------------------------------------------
// 6) Fused residual add + LayerNorm (biased var, eps=1e-6). In-place safe.
// ---------------------------------------------------------------------------
__global__ __launch_bounds__(256)
void add_ln_kernel(const float* __restrict__ a,
                   const float* __restrict__ x,
                   const float* __restrict__ gamma,
                   const float* __restrict__ beta,
                   float* __restrict__ out)
{
    __shared__ float red[256];
    const int row = blockIdx.x;
    const int t   = threadIdx.x;
    const float* ap = a + (size_t)row * D_MODEL;
    const float* xp = x + (size_t)row * D_MODEL;

    float v[4];
    float s = 0.0f;
#pragma unroll
    for (int i = 0; i < 4; ++i) {
        v[i] = ap[t + i * 256] + xp[t + i * 256];
        s += v[i];
    }
    red[t] = s;
    __syncthreads();
#pragma unroll
    for (int w = 128; w > 0; w >>= 1) {
        if (t < w) red[t] += red[t + w];
        __syncthreads();
    }
    const float mean = red[0] * (1.0f / D_MODEL);
    __syncthreads();

    float sq = 0.0f;
#pragma unroll
    for (int i = 0; i < 4; ++i) {
        const float d = v[i] - mean;
        sq += d * d;
    }
    red[t] = sq;
    __syncthreads();
#pragma unroll
    for (int w = 128; w > 0; w >>= 1) {
        if (t < w) red[t] += red[t + w];
        __syncthreads();
    }
    const float inv = rsqrtf(red[0] * (1.0f / D_MODEL) + 1e-6f);
#pragma unroll
    for (int i = 0; i < 4; ++i) {
        const int c = t + i * 256;
        out[(size_t)row * D_MODEL + c] = (v[i] - mean) * inv * gamma[c] + beta[c];
    }
}

// ---------------------------------------------------------------------------
// Host-side orchestration
// ---------------------------------------------------------------------------
extern "C" void kernel_launch(void* const* d_in, const int* in_sizes, int n_in,
                              void* d_out, int out_size, void* d_ws, size_t ws_size,
                              hipStream_t stream)
{
    (void)in_sizes; (void)n_in; (void)out_size; (void)ws_size;

    const int*   tok  = (const int*)d_in[0];
    const float* enc  = (const float*)d_in[1];   // [B,S,D]
    const float* pad  = (const float*)d_in[2];   // [B,1,1,S]
    const float* look = (const float*)d_in[3];   // [1,1,S,S]
    const float* emb  = (const float*)d_in[4];   // [V+1,D]

    float* out = (float*)d_out;
    float* ws  = (float*)d_ws;

    const size_t XSZ = (size_t)NTOK * D_MODEL;       // 2,097,152 floats
    const size_t ASZ = (size_t)BH * SEQ * SEQ;       // 16,777,216 floats

    float* x  = ws;              // activations     [NTOK, D]
    float* Qb = x  + XSZ;
    float* Kb = Qb + XSZ;
    float* Vb = Kb + XSZ;
    float* Ab = Vb + XSZ;        // concat attention output
    float* Pb = Ab + XSZ;        // proj / ffn-out scratch
    float* Fm = Pb + XSZ;        // ffn hidden      [NTOK, DFF]

    embed_kernel<<<NTOK, 256, 0, stream>>>(tok, emb, x);

    const dim3 gProj(D_MODEL / BN, NTOK / BM);       // (16, 16)
    const dim3 gFfn1(DFF / BN,     NTOK / BM);       // (64, 16)
    const dim3 gScore(SEQ / BN, SEQ / BM, BH);       // (8, 4, 64)
    const dim3 gAv(1, SEQ / BM, BH);                 // (1, 4, 64)

    for (int L = 0; L < 6; ++L) {
        const int base = 5 + L * 26;   // sorted pytree leaves per layer:
        // +0 ffn.d1.b +1 ffn.d1.w +2 ffn.d2.b +3 ffn.d2.w
        // +4 ln1.b +5 ln1.g +6 ln2.b +7 ln2.g +8 ln3.b +9 ln3.g
        // +10 mha1.k.b +11 mha1.k.w +12 mha1.o.b +13 mha1.o.w
        // +14 mha1.q.b +15 mha1.q.w +16 mha1.v.b +17 mha1.v.w
        // +18..25 mha2 in the same k/o/q/v order
        auto P = [&](int o) { return (const float*)d_in[base + o]; };

        // ---- self-attention (padding mask, per reference) ----
        gemm_bias_kernel<0><<<gProj, 256, 0, stream>>>(x, P(15), P(14), Qb, NTOK, D_MODEL, D_MODEL);
        gemm_bias_kernel<0><<<gProj, 256, 0, stream>>>(x, P(11), P(10), Kb, NTOK, D_MODEL, D_MODEL);
        gemm_bias_kernel<0><<<gProj, 256, 0, stream>>>(x, P(17), P(16), Vb, NTOK, D_MODEL, D_MODEL);

        float* probs1 = out + XSZ + (size_t)L * 2 * ASZ;
        attn_scores_kernel<<<gScore, 256, 0, stream>>>(Qb, Kb, pad, 0, probs1);
        softmax_kernel<<<BH * SEQ, 256, 0, stream>>>(probs1);
        attn_v_kernel<<<gAv, 256, 0, stream>>>(probs1, Vb, Ab);

        gemm_bias_kernel<0><<<gProj, 256, 0, stream>>>(Ab, P(13), P(12), Pb, NTOK, D_MODEL, D_MODEL);
        add_ln_kernel<<<NTOK, 256, 0, stream>>>(Pb, x, P(5), P(4), x);

        // ---- cross-attention (lookahead mask, per reference) ----
        gemm_bias_kernel<0><<<gProj, 256, 0, stream>>>(x,   P(23), P(22), Qb, NTOK, D_MODEL, D_MODEL);
        gemm_bias_kernel<0><<<gProj, 256, 0, stream>>>(enc, P(19), P(18), Kb, NTOK, D_MODEL, D_MODEL);
        gemm_bias_kernel<0><<<gProj, 256, 0, stream>>>(enc, P(25), P(24), Vb, NTOK, D_MODEL, D_MODEL);

        float* probs2 = probs1 + ASZ;
        attn_scores_kernel<<<gScore, 256, 0, stream>>>(Qb, Kb, look, 1, probs2);
        softmax_kernel<<<BH * SEQ, 256, 0, stream>>>(probs2);
        attn_v_kernel<<<gAv, 256, 0, stream>>>(probs2, Vb, Ab);

        gemm_bias_kernel<0><<<gProj, 256, 0, stream>>>(Ab, P(21), P(20), Pb, NTOK, D_MODEL, D_MODEL);
        add_ln_kernel<<<NTOK, 256, 0, stream>>>(Pb, x, P(7), P(6), x);

        // ---- FFN ----
        gemm_bias_kernel<1><<<gFfn1, 256, 0, stream>>>(x,  P(1), P(0), Fm, NTOK, DFF,     D_MODEL);
        gemm_bias_kernel<0><<<gProj, 256, 0, stream>>>(Fm, P(3), P(2), Pb, NTOK, D_MODEL, DFF);

        float* xdst = (L == 5) ? out : x;   // last layer writes final x to d_out
        add_ln_kernel<<<NTOK, 256, 0, stream>>>(Pb, x, P(9), P(8), xdst);
    }
}
